// SA_global_Swin_40415642255816
// MI455X (gfx1250) — compile-verified, matrix operand order checked by reading, not run on hardware
//
#include <hip/hip_runtime.h>
#include <hip/hip_bf16.h>
#include <math.h>

// ---------------- types for WMMA ----------------
typedef __attribute__((ext_vector_type(16))) __bf16         v16bf;
typedef __attribute__((ext_vector_type(16))) unsigned short v16us;
typedef __attribute__((ext_vector_type(8)))  unsigned int   v8u;
typedef __attribute__((ext_vector_type(8)))  float          v8f;

// ---------------- problem constants ----------------
static constexpr int HEADS   = 4;
static constexpr int HD      = 32;     // head dim
static constexpr int CHN     = 128;    // DIM_H (per-half channels) == WMMA K total
static constexpr int OC      = 384;    // q,k,v output channels
static constexpr int IMG     = 192;
static constexpr int TIN     = 14;     // interior tile (even -> windows align)
static constexpr int NPX     = 256;    // 16x16 region pixels (interior + 1px halo)
static constexpr int XS_STRIDE  = 136; // bf16 elems per pixel row of x (128 + pad)
static constexpr int QKV_STRIDE = 392; // bf16 elems per pixel row of qkv (384 + pad)
static constexpr int NTILES_X   = (IMG + TIN - 1) / TIN;   // 14

static constexpr size_t SMEM_BYTES =
    (size_t)(NPX * XS_STRIDE + NPX * QKV_STRIDE) * 2      // xs + qkvs (bf16)
    + (size_t)(CHN * 9 + CHN + HEADS * 16) * 4;           // lepe_w, lepe_b, bias

__device__ __forceinline__ unsigned short f2bf(float f) {
  unsigned int u = __float_as_uint(f);
  unsigned int r = u + 0x7FFFu + ((u >> 16) & 1u);   // round-to-nearest-even
  return (unsigned short)(r >> 16);
}
__device__ __forceinline__ float bf2f(unsigned short h) {
  return __uint_as_float(((unsigned int)h) << 16);
}

__global__ __launch_bounds__(256)
void swin_axial_fused(const float* __restrict__ x,
                      const float* __restrict__ qkv_w,
                      const float* __restrict__ lepe_w,
                      const float* __restrict__ lepe_b,
                      const float* __restrict__ bias_table,
                      const int*   __restrict__ rel_idx,
                      float* __restrict__ out) {
  extern __shared__ unsigned short smem_us[];
  unsigned short* xs   = smem_us;                       // [256][136] bf16 x region
  unsigned short* qkvs = smem_us + NPX * XS_STRIDE;     // [256][392] bf16 q|k|v, head-deinterleaved
  float* lepe_ws = (float*)(qkvs + NPX * QKV_STRIDE);   // [128*9]
  float* lepe_bs = lepe_ws + CHN * 9;                   // [128]
  float* bias_s  = lepe_bs + CHN;                       // [head][i][j] = 4*16

  const int tid  = threadIdx.x;
  const int tx   = blockIdx.x, ty = blockIdx.y, nb = blockIdx.z;  // nb: 0..7
  const int half = nb >> 2, bb = nb & 3;                          // n = half*B + b
  const int gx0  = tx * TIN - 1, gy0 = ty * TIN - 1;              // region origin (with halo)
  const float* xin = x + (size_t)(bb * 256 + half * 128) * IMG * IMG;

  // ---- preload small tables into LDS ----
  for (int i = tid; i < CHN * 9; i += 256) lepe_ws[i] = lepe_w[i];
  if (tid < CHN) lepe_bs[tid] = lepe_b[tid];
  if (tid < HEADS * 16) {
    int head = tid >> 4, ij = tid & 15;
    bias_s[tid] = bias_table[rel_idx[ij] * HEADS + head];
  }

  // ---- Phase 1: load 16x16 x region -> bf16 LDS (zero-fill outside image) ----
  {
    const int rp = tid;                      // 256 threads <-> 256 region pixels
    const int ry = rp >> 4, rx = rp & 15;
    const int gy = gy0 + ry, gx = gx0 + rx;
    const bool inb = (gy >= 0) & (gy < IMG) & (gx >= 0) & (gx < IMG);
    const float* src = xin + (size_t)gy * IMG + gx;
    unsigned short* dst = xs + rp * XS_STRIDE;
#pragma unroll 4
    for (int k = 0; k < CHN; ++k) {
      float v = inb ? src[(size_t)k * IMG * IMG] : 0.0f;
      dst[k] = f2bf(v);
    }
  }
  __syncthreads();

  // ---- Phase 2: qkv GEMM  out[384 x 256px] = W[384x128] @ x[128 x 256px] ----
  // bf16 WMMA 16x16x32, fp32 accumulate. 24 M-tiles / 8 waves = 3 per wave.
  const int wave  = tid >> 5;
  const int lane  = tid & 31;
  const int lhalf = lane >> 4;     // lane group 0/1
  const int l15   = lane & 15;

  for (int mi = 0; mi < 3; ++mi) {
    const int m0 = wave * 3 + mi;                 // M-tile 0..23
    v8f acc[16];
#pragma unroll
    for (int nn = 0; nn < 16; ++nn) {
#pragma unroll
      for (int r = 0; r < 8; ++r) acc[nn][r] = 0.0f;
    }

    for (int ks = 0; ks < 4; ++ks) {
      // A-fragment (weights): lane<16 holds M=l15, K = kb+{0..7,16..23};
      //                       lane>=16 holds M=l15, K = kb+{8..15,24..31}
      const int oc = m0 * 16 + l15;
      const int kb = ks * 32 + lhalf * 8;
      const float4* wp4 = (const float4*)(qkv_w + oc * CHN + kb);
      const float4 f0 = wp4[0], f1 = wp4[1], f2 = wp4[4], f3 = wp4[5];
      v16us abits;
      abits[0]  = f2bf(f0.x); abits[1]  = f2bf(f0.y); abits[2]  = f2bf(f0.z); abits[3]  = f2bf(f0.w);
      abits[4]  = f2bf(f1.x); abits[5]  = f2bf(f1.y); abits[6]  = f2bf(f1.z); abits[7]  = f2bf(f1.w);
      abits[8]  = f2bf(f2.x); abits[9]  = f2bf(f2.y); abits[10] = f2bf(f2.z); abits[11] = f2bf(f2.w);
      abits[12] = f2bf(f3.x); abits[13] = f2bf(f3.y); abits[14] = f2bf(f3.z); abits[15] = f2bf(f3.w);
      const v16bf afrag = __builtin_bit_cast(v16bf, abits);

#pragma unroll
      for (int nn = 0; nn < 16; ++nn) {
        // B-fragment (x): lane = pixel column, same K split as A (mirrored layout)
        const int p = nn * 16 + l15;
        const uint4* bp4 = (const uint4*)(xs + p * XS_STRIDE + kb);
        const uint4 b0 = bp4[0];   // K rel 0..7
        const uint4 b1 = bp4[2];   // K rel 16..23
        v8u bu;
        bu[0] = b0.x; bu[1] = b0.y; bu[2] = b0.z; bu[3] = b0.w;
        bu[4] = b1.x; bu[5] = b1.y; bu[6] = b1.z; bu[7] = b1.w;
        const v16bf bfrag = __builtin_bit_cast(v16bf, bu);

        acc[nn] = __builtin_amdgcn_wmma_f32_16x16x32_bf16(
            false, afrag, false, bfrag, (short)0, acc[nn], false, false);
      }
    }

    // store C/D -> LDS qkv, de-interleaving ch = c*4+head into [part][head][c]
    // D layout: lane<16: pixel N=l15, rows M=r; lane>=16: pixel N=l15, rows M=8+r
    const int ocbase = m0 * 16 + lhalf * 8;
#pragma unroll
    for (int nn = 0; nn < 16; ++nn) {
      const int p = nn * 16 + l15;
      unsigned short* qrow = qkvs + p * QKV_STRIDE;
#pragma unroll
      for (int r = 0; r < 8; ++r) {
        const int oc   = ocbase + r;
        const int part = oc >> 7;       // 0=q, 1=k, 2=v
        const int wi   = oc & 127;      // wi = c*4 + head
        qrow[part * 128 + (wi & 3) * HD + (wi >> 2)] = f2bf(acc[nn][r]);
      }
    }
  }
  __syncthreads();

  // ---- Phase 3: windowed attention (T=4) + bias + softmax + lepe ----
  const float scale = 0.17677669529663687f;   // 32^-0.5
  for (int t = tid; t < HEADS * TIN * TIN; t += 256) {
    const int head = t / (TIN * TIN);
    const int rr   = t - head * (TIN * TIN);
    const int iy   = rr / TIN, ix = rr - (rr / TIN) * TIN;
    const int gy   = ty * TIN + iy, gx = tx * TIN + ix;
    if (gy >= IMG || gx >= IMG) continue;

    const int ry = iy + 1, rx = ix + 1;       // region coords
    const int rp = ry * 16 + rx;
    const int ti = (iy & 1) * 2 + (ix & 1);   // token index in window
    const int wry = (iy & ~1) + 1, wrx = (ix & ~1) + 1;

    int rpj[4];
#pragma unroll
    for (int j = 0; j < 4; ++j)
      rpj[j] = (wry + (j >> 1)) * 16 + (wrx + (j & 1));

    const unsigned short* qv = qkvs + rp * QKV_STRIDE + head * HD;

    // scores + bias
    float s[4];
#pragma unroll
    for (int j = 0; j < 4; ++j) {
      const unsigned short* kv = qkvs + rpj[j] * QKV_STRIDE + 128 + head * HD;
      float d = 0.0f;
#pragma unroll 8
      for (int c = 0; c < HD; ++c) d += bf2f(qv[c]) * bf2f(kv[c]);
      s[j] = d * scale + bias_s[head * 16 + ti * 4 + j];
    }
    const float mx = fmaxf(fmaxf(s[0], s[1]), fmaxf(s[2], s[3]));
    float e[4], sum = 0.0f;
#pragma unroll
    for (int j = 0; j < 4; ++j) { e[j] = __expf(s[j] - mx); sum += e[j]; }
    const float inv = 1.0f / sum;

    // out = attn@v + lepe (depthwise 3x3 on v, zero-padded via zero x halo)
#pragma unroll 4
    for (int c = 0; c < HD; ++c) {
      float o = 0.0f;
#pragma unroll
      for (int j = 0; j < 4; ++j)
        o += e[j] * bf2f(qkvs[rpj[j] * QKV_STRIDE + 256 + head * HD + c]);
      o *= inv;

      const int ch = c * 4 + head;
      float lp = lepe_bs[ch];
#pragma unroll
      for (int dy = -1; dy <= 1; ++dy) {
#pragma unroll
        for (int dx = -1; dx <= 1; ++dx) {
          const float wv = lepe_ws[ch * 9 + (dy + 1) * 3 + (dx + 1)];
          lp += wv * bf2f(qkvs[((ry + dy) * 16 + (rx + dx)) * QKV_STRIDE + 256 + head * HD + c]);
        }
      }
      out[((size_t)(bb * 256 + half * 128 + ch) * IMG + gy) * IMG + gx] = o + lp;
    }
  }
}

extern "C" void kernel_launch(void* const* d_in, const int* in_sizes, int n_in,
                              void* d_out, int out_size, void* d_ws, size_t ws_size,
                              hipStream_t stream) {
  (void)in_sizes; (void)n_in; (void)out_size; (void)d_ws; (void)ws_size;
  const float* x          = (const float*)d_in[0];
  const float* qkv_w      = (const float*)d_in[1];
  const float* lepe_w     = (const float*)d_in[2];
  const float* lepe_b     = (const float*)d_in[3];
  const float* bias_table = (const float*)d_in[4];
  const int*   rel_idx    = (const int*)d_in[5];
  float* out = (float*)d_out;

  dim3 grid(NTILES_X, NTILES_X, 8);   // 14 x 14 tiles, 8 logical batches (2 halves x B)
  swin_axial_fused<<<grid, 256, SMEM_BYTES, stream>>>(
      x, qkv_w, lepe_w, lepe_b, bias_table, rel_idx, out);
}
